// SE3ScoreModel_78082505441913
// MI455X (gfx1250) — matrix operand order repaired and checked by reading, not compiled, not run on hardware
//
#include <hip/hip_runtime.h>
#include <math.h>

typedef __attribute__((ext_vector_type(16))) _Float16 v16h;
typedef __attribute__((ext_vector_type(8)))  _Float16 h8;
typedef __attribute__((ext_vector_type(8)))  float    v8f;

#define G_   64
#define L_   256
#define K_   32
#define N_   (G_ * L_)       // 16384 nodes
#define E_   (N_ * K_)       // 524288 edges
#define HID_ 32
#define NB_  10

__device__ __constant__ float C_S       = 0.3826834323650898f;   // sin(pi/8)
__device__ __constant__ float C_X       = 0.9238795325112867f;   // cos(pi/8)
__device__ __constant__ float SQ3       = 1.7320508075688772f;
__device__ __constant__ float INV_SK    = 0.17677669529663687f;  // 1/sqrt(32)
__device__ __constant__ float INV_SH    = 0.17677669529663687f;  // 1/sqrt(HID)
__device__ __constant__ float INV_SNB   = 0.31622776601683794f;  // 1/sqrt(10)
__device__ __constant__ float INV_S2H   = 0.125f;                // 1/sqrt(64)
__device__ __constant__ float INV_STEP  = 1.8f;                  // 9/5
__device__ __constant__ float EMB_NORM  = 0.8928571428571429f;   // 1/1.12

// ---------------------------------------------------------------------------
// per-wave LDS fence: DS ops are in-order per wave; wait for stores to land
// before dependent loads (RAW). WAR is implicitly safe (in-order DS).
// ---------------------------------------------------------------------------
static __device__ inline void lds_fence() {
  __builtin_amdgcn_wave_barrier();
  asm volatile("s_wait_dscnt 0" ::: "memory");
  __builtin_amdgcn_wave_barrier();
}

// fast silu: v_exp + v_rcp instead of IEEE divide (v_div_scale/fmas chains)
static __device__ inline float fast_silu(float x) {
  return x * __builtin_amdgcn_rcpf(1.0f + __expf(-x));
}

static __device__ inline float wave_sum(float v) {
#pragma unroll
  for (int off = 16; off > 0; off >>= 1) v += __shfl_xor(v, off, 32);
  return v;
}

// ---------------------------------------------------------------------------
// WMMA helpers (V_WMMA_F32_16X16X32_F16). CDNA5 wave32 layouts (ISA 7.12.2):
//  A (16x32 f16): lane L holds row M=L&15; kb = (L&16)?8:0;
//                 v16h elem i -> K=kb+i (i<8), K=16+kb+(i-8) (i>=8).
//  B (32x16 f16): lane L holds col N=L&15; same K mapping as A (kb=(L&16)?8:0).
//  C/D (16x16 f32): VGPR r, lane L -> M = r + 8*(L>>4), N = L&15.
// ---------------------------------------------------------------------------
static __device__ inline v8f wmma_f16(v16h a, v16h b, v8f c) {
  return __builtin_amdgcn_wmma_f32_16x16x32_f16(false, a, false, b,
                                                (short)0, c, false, false);
}

// A fragment from row-major f16 LDS buffer buf[rows][32] (rows 64B aligned).
// Each lane's two K-chunks are contiguous -> 2x ds_load_b128.
static __device__ inline v16h load_a_frag(const _Float16* buf, int rowBase,
                                          int lane) {
  const int m  = lane & 15;
  const int kb = (lane & 16) ? 8 : 0;
  const _Float16* row = buf + (size_t)(rowBase + m) * 32;
  const h8 lo = *(const h8*)(row + kb);        // K = kb .. kb+7
  const h8 hi = *(const h8*)(row + 16 + kb);   // K = 16+kb .. 16+kb+7
  v16h a;
#pragma unroll
  for (int i = 0; i < 8; ++i) {
    a[i]     = lo[i];
    a[8 + i] = hi[i];
  }
  return a;
}

// Build B fragment from a row-major f32 weight matrix W[kmax][ldw], scaled.
static __device__ inline v16h build_b_frag(const float* W, int ldw, int ncols,
                                           int colBase, int kmax, float scale,
                                           int lane) {
  const int n  = (lane & 15) + colBase;
  const int kb = (lane & 16) ? 8 : 0;
  v16h b;
#pragma unroll
  for (int i = 0; i < 16; ++i) {
    const int k = (i < 8) ? kb + i : 16 + kb + (i - 8);
    const float v = (k < kmax && n < ncols) ? W[(size_t)k * ldw + n] * scale : 0.0f;
    b[i] = (_Float16)v;
  }
  return b;
}

// silu(D) -> f16 LDS buffer buf[16][32], writing the N-half `half_n`.
static __device__ inline void store_d_silu(_Float16* buf, v8f c, int half_n,
                                           int lane) {
  const int n  = (lane & 15) + 16 * half_n;
  const int mo = (lane & 16) ? 8 : 0;
#pragma unroll
  for (int r = 0; r < 8; ++r) {
    buf[(size_t)(r + mo) * 32 + n] = (_Float16)fast_silu(c[r]);
  }
}

// Gaussian radial basis -> one padded 32-half LDS row (4x ds_store_b128).
static __device__ inline void store_emb_row(_Float16* row, float r) {
  const float rs = r * INV_STEP;
  h8 chunk[4];
#pragma unroll
  for (int j = 0; j < 32; ++j) {
    float v = 0.0f;
    if (j < NB_) {
      const float d = rs - (float)j;
      v = __expf(-d * d) * EMB_NORM;
    }
    chunk[j >> 3][j & 7] = (_Float16)v;
  }
#pragma unroll
  for (int q = 0; q < 4; ++q) *((h8*)row + q) = chunk[q];
}

// ===========================================================================
// Kernel E1: one wave per node (32 contiguous edges). Geometry + MLP1 (WMMA)
// + masked message formation + in-wave segment sum -> nms[N], nmv[N][3].
// Also stores r[E], sh1[E][3] for pass 2.
// ===========================================================================
__global__ __launch_bounds__(256) void e1_kernel(
    const float* __restrict__ coords, const float* __restrict__ t,
    const int* __restrict__ edge_src, const float* __restrict__ edge_mask,
    const float* __restrict__ lin1_w, const float* __restrict__ fc1_w0,
    const float* __restrict__ fc1_w1, const float* __restrict__ fc1_w2,
    float* __restrict__ r_ws, float* __restrict__ sh1_ws,
    float* __restrict__ nms, float* __restrict__ nmv) {
  __shared__ _Float16 embLds[8][32][32];   // padded Gaussian basis per edge
  __shared__ _Float16 hLds[8][16][32];     // layer transpose restage
  __shared__ float    w1Lds[8][32][2];     // MLP1 output per edge

  const int wave = threadIdx.x >> 5;
  const int lane = threadIdx.x & 31;
  const int node = blockIdx.x * 8 + wave;
  const int e    = node * K_ + lane;

  // ---- geometry + radial embedding (lane = edge) ----
  const int   src = edge_src[e];
  const float em  = edge_mask[e];
  const float vx = coords[node * 3 + 0] - coords[src * 3 + 0];
  const float vy = coords[node * 3 + 1] - coords[src * 3 + 1];
  const float vz = coords[node * 3 + 2] - coords[src * 3 + 2];
  const float d2    = vx * vx + vy * vy + vz * vz + 1e-12f;
  const float inv_r = __builtin_amdgcn_rsqf(d2);   // 1/sqrt(d2)
  const float r     = d2 * inv_r;                  // sqrt(d2)
  const float s3r = SQ3 * inv_r;
  const float sh0 = s3r * vx, sh1v = s3r * vy, sh2 = s3r * vz;
  r_ws[e] = r;
  sh1_ws[e * 3 + 0] = sh0;
  sh1_ws[e * 3 + 1] = sh1v;
  sh1_ws[e * 3 + 2] = sh2;
  store_emb_row(&embLds[wave][lane][0], r);
  lds_fence();

  // ---- MLP1 weight fragments (f32 global -> f16 frag, pre-scaled) ----
  const v16h b0a = build_b_frag(fc1_w0, HID_, HID_, 0, NB_, INV_SNB, lane);
  const v16h b0b = build_b_frag(fc1_w0, HID_, HID_, 16, NB_, INV_SNB, lane);
  const v16h b1a = build_b_frag(fc1_w1, HID_, HID_, 0, HID_, INV_SH, lane);
  const v16h b1b = build_b_frag(fc1_w1, HID_, HID_, 16, HID_, INV_SH, lane);
  const v16h b2  = build_b_frag(fc1_w2, 2, 2, 0, HID_, INV_SH, lane);

#pragma unroll
  for (int tile = 0; tile < 2; ++tile) {
    // layer 1: emb @ W0
    v16h a = load_a_frag(&embLds[wave][0][0], tile * 16, lane);
    v8f c0 = {}, c1 = {};
    c0 = wmma_f16(a, b0a, c0);
    c1 = wmma_f16(a, b0b, c1);
    store_d_silu(&hLds[wave][0][0], c0, 0, lane);
    store_d_silu(&hLds[wave][0][0], c1, 1, lane);
    lds_fence();
    // layer 2: h @ W1
    a = load_a_frag(&hLds[wave][0][0], 0, lane);
    v8f d0 = {}, d1 = {};
    d0 = wmma_f16(a, b1a, d0);
    d1 = wmma_f16(a, b1b, d1);
    store_d_silu(&hLds[wave][0][0], d0, 0, lane);
    store_d_silu(&hLds[wave][0][0], d1, 1, lane);
    lds_fence();
    // layer 3: h @ W2 (2 valid columns)
    a = load_a_frag(&hLds[wave][0][0], 0, lane);
    v8f w = {};
    w = wmma_f16(a, b2, w);
    {
      const int n  = lane & 15;
      const int mo = (lane & 16) ? 8 : 0;
      if (n < 2) {
#pragma unroll
        for (int rr = 0; rr < 8; ++rr)
          w1Lds[wave][tile * 16 + mo + rr][n] = w[rr];
      }
    }
    lds_fence();
  }

  // ---- messages + segment sum over this node's 32 edges ----
  const float asrc = t[src >> 8];  // batch(src) = src / L_
  const float xs   = asrc * lin1_w[0];
  const float w10  = w1Lds[wave][lane][0];
  const float w11  = w1Lds[wave][lane][1];
  float ms  = w10 * xs * em;
  const float cf = w11 * xs * em;
  float mv0 = cf * sh0, mv1 = cf * sh1v, mv2 = cf * sh2;
  ms  = wave_sum(ms);
  mv0 = wave_sum(mv0);
  mv1 = wave_sum(mv1);
  mv2 = wave_sum(mv2);
  if (lane == 0) {
    nms[node]         = ms * INV_SK;
    nmv[node * 3 + 0] = mv0 * INV_SK;
    nmv[node * 3 + 1] = mv1 * INV_SK;
    nmv[node * 3 + 2] = mv2 * INV_SK;
  }
}

// ===========================================================================
// Kernel N: one wave per 16-node tile. Elementwise h_s / h_v, then WMMA for
// s1 = h_s@lin1s_w and v1_c = h_v_c@lin1v_w (4 GEMMs, 8 WMMAs/wave). sc_v.
// ===========================================================================
__global__ __launch_bounds__(256) void node_kernel(
    const float* __restrict__ t, const float* __restrict__ sc1_w,
    const float* __restrict__ conv1_l2s, const float* __restrict__ conv1_l2v,
    const float* __restrict__ sc2_w, const float* __restrict__ lin1s_w,
    const float* __restrict__ lin1v_w, const float* __restrict__ nms,
    const float* __restrict__ nmv, float* __restrict__ scv,
    float* __restrict__ s1_ws, float* __restrict__ v1_ws) {
  __shared__ _Float16 hsLds[8][16][32];
  __shared__ _Float16 hvLds[8][3][16][32];

  const int wave = threadIdx.x >> 5;
  const int lane = threadIdx.x & 31;
  const int nodeBase = (blockIdx.x * 8 + wave) * 16;

  // ---- elementwise h_s, h_v into LDS ----
#pragma unroll
  for (int it = 0; it < 16; ++it) {
    const int idx  = it * 32 + lane;
    const int m    = idx >> 5;
    const int u    = idx & 31;
    const int node = nodeBase + m;
    const float a    = t[node >> 8];
    const float nmsv = nms[node];
    const float hs = fast_silu(C_S * a * sc1_w[u] +
                               C_X * nmsv * a * conv1_l2s[u]);
    hsLds[wave][m][u] = (_Float16)hs;
#pragma unroll
    for (int c = 0; c < 3; ++c) {
      const float hv = fast_silu(C_X * nmv[node * 3 + c] * a * conv1_l2v[u]);
      hvLds[wave][c][m][u] = (_Float16)hv;
    }
  }
  lds_fence();

  // ---- sc_v (small reductions from LDS) ----
  for (int base = lane; base < 48; base += 32) {
    const int m = base / 3, c = base % 3;
    const int node = nodeBase + m;
    float s = 0.0f;
#pragma unroll
    for (int u = 0; u < HID_; ++u)
      s += (float)hvLds[wave][c][m][u] * sc2_w[u];
    scv[node * 3 + c] = s * t[node >> 8] * INV_SH;
  }

  // ---- WMMA: s1 and v1 ----
  const v16h bsa = build_b_frag(lin1s_w, HID_, HID_, 0, HID_, INV_SH, lane);
  const v16h bsb = build_b_frag(lin1s_w, HID_, HID_, 16, HID_, INV_SH, lane);
  const v16h bva = build_b_frag(lin1v_w, HID_, HID_, 0, HID_, INV_SH, lane);
  const v16h bvb = build_b_frag(lin1v_w, HID_, HID_, 16, HID_, INV_SH, lane);

  const int mo = (lane & 16) ? 8 : 0;
  {
    const v16h a = load_a_frag(&hsLds[wave][0][0], 0, lane);
    v8f c0 = {}, c1 = {};
    c0 = wmma_f16(a, bsa, c0);
    c1 = wmma_f16(a, bsb, c1);
#pragma unroll
    for (int h = 0; h < 2; ++h) {
      const v8f& d = h ? c1 : c0;
      const int n = (lane & 15) + 16 * h;
#pragma unroll
      for (int rr = 0; rr < 8; ++rr) {
        const int node = nodeBase + mo + rr;
        s1_ws[(size_t)node * HID_ + n] = d[rr] * t[node >> 8];
      }
    }
  }
#pragma unroll
  for (int c = 0; c < 3; ++c) {
    const v16h a = load_a_frag(&hvLds[wave][c][0][0], 0, lane);
    v8f c0 = {}, c1 = {};
    c0 = wmma_f16(a, bva, c0);
    c1 = wmma_f16(a, bvb, c1);
#pragma unroll
    for (int h = 0; h < 2; ++h) {
      const v8f& d = h ? c1 : c0;
      const int n = (lane & 15) + 16 * h;
#pragma unroll
      for (int rr = 0; rr < 8; ++rr) {
        const int node = nodeBase + mo + rr;
        v1_ws[(size_t)node * 96 + n * 3 + c] = d[rr] * t[node >> 8];
      }
    }
  }
}

// ===========================================================================
// Kernel E2: one wave per node. Rebuild emb from r, MLP2 via WMMA (8/tile),
// then lane-per-channel gather of s1[src]/v1[src] (L2 resident), m1/m2
// accumulation, shuffle reduce, final output.
// ===========================================================================
__global__ __launch_bounds__(256) void e2_kernel(
    const float* __restrict__ t, const int* __restrict__ edge_src,
    const float* __restrict__ edge_mask, const float* __restrict__ fc2_w0,
    const float* __restrict__ fc2_w1, const float* __restrict__ fc2_w2,
    const float* __restrict__ conv2_l2a, const float* __restrict__ conv2_l2b,
    const float* __restrict__ r_ws, const float* __restrict__ sh1_ws,
    const float* __restrict__ scv, const float* __restrict__ s1_ws,
    const float* __restrict__ v1_ws, float* __restrict__ out) {
  __shared__ _Float16 embLds[8][32][32];
  __shared__ _Float16 hLds[8][16][32];
  __shared__ _Float16 w2Lds[8][32][64];
  __shared__ int      srcLds[8][32];
  __shared__ float    emLds[8][32];
  __shared__ float    shLds[8][32][3];

  const int wave = threadIdx.x >> 5;
  const int lane = threadIdx.x & 31;
  const int node = blockIdx.x * 8 + wave;
  const int e    = node * K_ + lane;

  // ---- stage per-edge data; rebuild Gaussian basis from r ----
  const float r = r_ws[e];
  srcLds[wave][lane] = edge_src[e];
  emLds[wave][lane]  = edge_mask[e];
#pragma unroll
  for (int c = 0; c < 3; ++c) shLds[wave][lane][c] = sh1_ws[e * 3 + c];
  store_emb_row(&embLds[wave][lane][0], r);
  lds_fence();

  // ---- MLP2 weight fragments ----
  const v16h b0a = build_b_frag(fc2_w0, HID_, HID_, 0, NB_, INV_SNB, lane);
  const v16h b0b = build_b_frag(fc2_w0, HID_, HID_, 16, NB_, INV_SNB, lane);
  const v16h b1a = build_b_frag(fc2_w1, HID_, HID_, 0, HID_, INV_SH, lane);
  const v16h b1b = build_b_frag(fc2_w1, HID_, HID_, 16, HID_, INV_SH, lane);
  v16h b2[4];
#pragma unroll
  for (int h = 0; h < 4; ++h)
    b2[h] = build_b_frag(fc2_w2, 64, 64, h * 16, HID_, INV_SH, lane);

#pragma unroll
  for (int tile = 0; tile < 2; ++tile) {
    v16h a = load_a_frag(&embLds[wave][0][0], tile * 16, lane);
    v8f c0 = {}, c1 = {};
    c0 = wmma_f16(a, b0a, c0);
    c1 = wmma_f16(a, b0b, c1);
    store_d_silu(&hLds[wave][0][0], c0, 0, lane);
    store_d_silu(&hLds[wave][0][0], c1, 1, lane);
    lds_fence();
    a = load_a_frag(&hLds[wave][0][0], 0, lane);
    v8f d0 = {}, d1 = {};
    d0 = wmma_f16(a, b1a, d0);
    d1 = wmma_f16(a, b1b, d1);
    store_d_silu(&hLds[wave][0][0], d0, 0, lane);
    store_d_silu(&hLds[wave][0][0], d1, 1, lane);
    lds_fence();
    a = load_a_frag(&hLds[wave][0][0], 0, lane);
    const int mo = (lane & 16) ? 8 : 0;
#pragma unroll
    for (int h = 0; h < 4; ++h) {
      v8f w = {};
      w = wmma_f16(a, b2[h], w);
      const int n = h * 16 + (lane & 15);
#pragma unroll
      for (int rr = 0; rr < 8; ++rr)
        w2Lds[wave][tile * 16 + mo + rr][n] = (_Float16)w[rr];
    }
    lds_fence();
  }

  // ---- second message pass: lane = channel u, loop over 32 edges ----
  const int u = lane;
  float a1x = 0.f, a1y = 0.f, a1z = 0.f;
  float a2x = 0.f, a2y = 0.f, a2z = 0.f;
#pragma unroll 4
  for (int ee = 0; ee < K_; ++ee) {
    const int   se  = srcLds[wave][ee];
    const float eme = emLds[wave][ee];
    const float wa  = (float)w2Lds[wave][ee][u];
    const float wb  = (float)w2Lds[wave][ee][32 + u];
    const float p   = wa * s1_ws[(size_t)se * HID_ + u] * eme;
    const float q   = wb * eme;
    const float* vp = v1_ws + (size_t)se * 96 + u * 3;
    a1x += p * shLds[wave][ee][0];
    a1y += p * shLds[wave][ee][1];
    a1z += p * shLds[wave][ee][2];
    a2x += q * vp[0];
    a2y += q * vp[1];
    a2z += q * vp[2];
  }
  const float l2a = conv2_l2a[u], l2b = conv2_l2b[u];
  float vc0 = wave_sum(a1x * l2a + a2x * l2b);
  float vc1 = wave_sum(a1y * l2a + a2y * l2b);
  float vc2 = wave_sum(a1z * l2a + a2z * l2b);
  if (lane == 0) {
    const float sc = INV_SK * t[node >> 8] * INV_S2H * C_X;
    out[node * 3 + 0] = C_S * scv[node * 3 + 0] + vc0 * sc;
    out[node * 3 + 1] = C_S * scv[node * 3 + 1] + vc1 * sc;
    out[node * 3 + 2] = C_S * scv[node * 3 + 2] + vc2 * sc;
  }
}

// ===========================================================================
extern "C" void kernel_launch(void* const* d_in, const int* in_sizes, int n_in,
                              void* d_out, int out_size, void* d_ws,
                              size_t ws_size, hipStream_t stream) {
  (void)in_sizes; (void)n_in; (void)out_size; (void)ws_size;
  const float* coords    = (const float*)d_in[0];
  const float* t         = (const float*)d_in[1];
  const int*   edge_src  = (const int*)d_in[3];
  const float* edge_mask = (const float*)d_in[5];
  const float* sc1_w     = (const float*)d_in[6];
  const float* lin1_w    = (const float*)d_in[7];
  const float* fc1_w0    = (const float*)d_in[8];
  const float* fc1_w1    = (const float*)d_in[9];
  const float* fc1_w2    = (const float*)d_in[10];
  const float* conv1_l2s = (const float*)d_in[11];
  const float* conv1_l2v = (const float*)d_in[12];
  const float* sc2_w     = (const float*)d_in[13];
  const float* lin1s_w   = (const float*)d_in[14];
  const float* lin1v_w   = (const float*)d_in[15];
  const float* fc2_w0    = (const float*)d_in[16];
  const float* fc2_w1    = (const float*)d_in[17];
  const float* fc2_w2    = (const float*)d_in[18];
  const float* conv2_l2a = (const float*)d_in[19];
  const float* conv2_l2b = (const float*)d_in[20];
  float* outp = (float*)d_out;

  // workspace layout (floats)
  float* W      = (float*)d_ws;
  float* r_ws   = W;                          // E
  float* sh1_ws = W + (size_t)E_;             // 3E
  float* nms    = W + (size_t)4 * E_;         // N
  float* nmv    = nms + N_;                   // 3N
  float* scv    = nmv + (size_t)3 * N_;       // 3N
  float* s1_ws  = scv + (size_t)3 * N_;       // 32N
  float* v1_ws  = s1_ws + (size_t)32 * N_;    // 96N

  e1_kernel<<<N_ / 8, 256, 0, stream>>>(coords, t, edge_src, edge_mask, lin1_w,
                                        fc1_w0, fc1_w1, fc1_w2, r_ws, sh1_ws,
                                        nms, nmv);
  node_kernel<<<N_ / 128, 256, 0, stream>>>(t, sc1_w, conv1_l2s, conv1_l2v,
                                            sc2_w, lin1s_w, lin1v_w, nms, nmv,
                                            scv, s1_ws, v1_ws);
  e2_kernel<<<N_ / 8, 256, 0, stream>>>(t, edge_src, edge_mask, fc2_w0, fc2_w1,
                                        fc2_w2, conv2_l2a, conv2_l2b, r_ws,
                                        sh1_ws, scv, s1_ws, v1_ws, outp);
}
